// GE_GNN_12180527252008
// MI455X (gfx1250) — matrix operand order repaired and checked by reading, not compiled, not run on hardware
//
#include <hip/hip_runtime.h>
#include <hip/hip_bf16.h>
#include <stdint.h>

#define NN 50000
#define NE 800000
#define HID 128
#define GDEPTH 3
#define NGRAPH 256
#define BN_EPS 1e-5f
#define KS 136   // padded LDS row stride in bf16 elements (even, 8B-aligned rows)

typedef __attribute__((ext_vector_type(16))) __bf16 v16bf;
typedef __attribute__((ext_vector_type(8)))  float  v8f;

union Frag { v16bf v; unsigned int w[8]; };

// tanh-approx GELU (jax.nn.gelu approximate=True), branchless:
// 0.5*x*(1+tanh(u)) = x * (1 - 1/(1+exp2(2*log2e*u))),  u = c0*(x + c1*x^3)
// exp2(+inf)->inf, rcp(inf)->0  and  exp2(-inf)->0, rcp(1)->1  give correct saturation.
__device__ __forceinline__ float gelu_t(float x) {
    const float c0 = 0.7978845608028654f, c1 = 0.044715f;
    const float two_log2e = 2.885390081777927f;     // 2/ln(2) scaled: 2*log2(e)
    float u = c0 * (x + c1 * x * x * x);
    float e = __builtin_amdgcn_exp2f(u * two_log2e);
    return x * (1.0f - __builtin_amdgcn_rcpf(e + 1.0f));
}

__device__ __forceinline__ unsigned short f2bf(float f) {
    unsigned int u = __float_as_uint(f);
    u += 0x7fffu + ((u >> 16) & 1u);   // round-to-nearest-even
    return (unsigned short)(u >> 16);
}

// ---- WMMA fragment loaders (ISA 7.12.2 wave32 layouts) ----------------------
// A: 16x32 bf16. lanes 0-15: M=lane, K = k0+{0..7} and k0+{16..23};
//                lanes 16-31: M=lane-16, K shifted by +8. Contiguous pairs -> b32/b128 loads.
__device__ __forceinline__ v16bf fragA(const unsigned short* sX, int m0, int k0, int lane) {
    Frag f;
    const int m  = m0 + (lane & 15);
    const int h8 = (lane >> 4) * 8;
    const unsigned int* p = (const unsigned int*)(sX + m * KS + k0 + h8);
    f.w[0] = p[0]; f.w[1] = p[1]; f.w[2] = p[2]; f.w[3] = p[3];
    const unsigned int* q = (const unsigned int*)(sX + m * KS + k0 + 16 + h8);
    f.w[4] = q[0]; f.w[5] = q[1]; f.w[6] = q[2]; f.w[7] = q[3];
    return f.v;
}
// B: 32x16 bf16. lane holds column n=lane&15; lanes 0-15 K=k0+{0..7},{16..23}, lanes 16-31 K shifted by +8.
// Weights staged TRANSPOSED in LDS (Wt[n][k]) so each lane's K run is contiguous.
__device__ __forceinline__ v16bf fragB(const unsigned short* sWt, int n0, int k0, int lane) {
    Frag f;
    const int n = n0 + (lane & 15);
    const unsigned int* p = (const unsigned int*)(sWt + n * KS + k0 + ((lane >> 4) << 3));
#pragma unroll
    for (int j = 0; j < 4; ++j) { f.w[j] = p[j]; f.w[4 + j] = p[8 + j]; }
    return f.v;
}

// ---- cooperative tile loaders (256 threads) --------------------------------
__device__ __forceinline__ void load_Wt(unsigned short* sWt, const float* W, int tid) {
    for (int i = tid; i < HID * 32; i += 256) {
        int k = i >> 5, n = (i & 31) * 4;
        const float4 v = *(const float4*)(W + (long long)k * HID + n);  // coalesced along n
        sWt[(n + 0) * KS + k] = f2bf(v.x);
        sWt[(n + 1) * KS + k] = f2bf(v.y);
        sWt[(n + 2) * KS + k] = f2bf(v.z);
        sWt[(n + 3) * KS + k] = f2bf(v.w);
    }
}

// mode 0: X = in0[row]            (plain rows)
// mode 1: X = scale*in0 + in1     (GIN node input: (1+eps)*nh + agg)
// mode 2: X = in0 + in1[src[row]] + in1[dst[row]]   (GIN edge input, fused gather)
__device__ __forceinline__ void load_X(unsigned short* sX, long long row0, long long rows,
                                       int mode, const float* in0, const float* in1,
                                       const int* src, const int* dst, float scale, int tid) {
    for (int i = tid; i < HID * 32; i += 256) {
        int r = i >> 5, c = (i & 31) * 4;
        long long row = row0 + r;
        float4 v = make_float4(0.f, 0.f, 0.f, 0.f);
        if (row < rows) {
            float4 a = *(const float4*)(in0 + row * HID + c);
            if (mode == 0) {
                v = a;
            } else if (mode == 1) {
                float4 g = *(const float4*)(in1 + row * HID + c);
                v = make_float4(scale * a.x + g.x, scale * a.y + g.y,
                                scale * a.z + g.z, scale * a.w + g.w);
            } else {
                long long s = src[row], d = dst[row];
                float4 bs = *(const float4*)(in1 + s * HID + c);
                float4 bd = *(const float4*)(in1 + d * HID + c);
                v = make_float4(a.x + bs.x + bd.x, a.y + bs.y + bd.y,
                                a.z + bs.z + bd.z, a.w + bs.w + bd.w);
            }
        }
        unsigned short* o = sX + r * KS + c;
        o[0] = f2bf(v.x); o[1] = f2bf(v.y); o[2] = f2bf(v.z); o[3] = f2bf(v.w);
    }
}

// ---- fused 2-layer MLP: out = gelu(X@W1+b1)@W2+b2  (X built per `mode`) ----
__global__ __launch_bounds__(256) void mlp128_fused(
    const float* __restrict__ in0, const float* __restrict__ in1,
    const int* __restrict__ src, const int* __restrict__ dst,
    const float* __restrict__ epsp, int mode,
    const float* __restrict__ W1, const float* __restrict__ b1,
    const float* __restrict__ W2, const float* __restrict__ b2,
    float* __restrict__ out, long long rows)
{
    __shared__ unsigned short sX [HID * KS];
    __shared__ unsigned short sW1t[HID * KS];
    __shared__ unsigned short sW2t[HID * KS];
    __shared__ unsigned short sH [HID * KS];
    const int tid = threadIdx.x;
    const long long row0 = (long long)blockIdx.x * HID;
    float scale = 1.0f;
    if (mode == 1) scale = 1.0f + epsp[0];

    load_X(sX, row0, rows, mode, in0, in1, src, dst, scale, tid);
    load_Wt(sW1t, W1, tid);
    load_Wt(sW2t, W2, tid);
    __syncthreads();

    const int lane  = tid & 31;
    const int m0    = (tid >> 5) * 16;   // 8 waves x 16-row stripes
    const int nlane = lane & 15;
    const int half  = lane >> 4;

    // layer 1 -> gelu -> sH (bf16)
#pragma unroll
    for (int nt = 0; nt < 8; ++nt) {
        v8f acc = {0.f, 0.f, 0.f, 0.f, 0.f, 0.f, 0.f, 0.f};
#pragma unroll
        for (int kt = 0; kt < 4; ++kt) {
            v16bf a = fragA(sX,  m0,      kt * 32, lane);
            v16bf b = fragB(sW1t, nt * 16, kt * 32, lane);
            acc = __builtin_amdgcn_wmma_f32_16x16x32_bf16(false, a, false, b,
                                                          (short)0, acc, false, false);
        }
        const int n = nt * 16 + nlane;
        const float bias = b1[n];
#pragma unroll
        for (int r = 0; r < 8; ++r)
            sH[(m0 + r + 8 * half) * KS + n] = f2bf(gelu_t(acc[r] + bias));
    }
    __syncthreads();   // (H is per-wave, but barrier is cheap insurance)

    // layer 2 -> global
#pragma unroll
    for (int nt = 0; nt < 8; ++nt) {
        v8f acc = {0.f, 0.f, 0.f, 0.f, 0.f, 0.f, 0.f, 0.f};
#pragma unroll
        for (int kt = 0; kt < 4; ++kt) {
            v16bf a = fragA(sH,  m0,      kt * 32, lane);
            v16bf b = fragB(sW2t, nt * 16, kt * 32, lane);
            acc = __builtin_amdgcn_wmma_f32_16x16x32_bf16(false, a, false, b,
                                                          (short)0, acc, false, false);
        }
        const int n = nt * 16 + nlane;
        const float bias = b2[n];
#pragma unroll
        for (int r = 0; r < 8; ++r) {
            long long row = row0 + m0 + r + 8 * half;
            if (row < rows) out[row * HID + n] = acc[r] + bias;
        }
    }
}

// ---- single 128x128 linear + fused BN batch-stat atomics (encoder layer 2) -
__global__ __launch_bounds__(256) void linear128_stats(
    const float* __restrict__ in0, const float* __restrict__ W, const float* __restrict__ b,
    float* __restrict__ out, float* __restrict__ stats, long long rows)
{
    __shared__ unsigned short sX [HID * KS];
    __shared__ unsigned short sWt[HID * KS];
    const int tid = threadIdx.x;
    const long long row0 = (long long)blockIdx.x * HID;

    load_X(sX, row0, rows, 0, in0, nullptr, nullptr, nullptr, 1.0f, tid);
    load_Wt(sWt, W, tid);
    __syncthreads();

    const int lane = tid & 31, m0 = (tid >> 5) * 16;
    const int nlane = lane & 15, half = lane >> 4;
#pragma unroll
    for (int nt = 0; nt < 8; ++nt) {
        v8f acc = {0.f, 0.f, 0.f, 0.f, 0.f, 0.f, 0.f, 0.f};
#pragma unroll
        for (int kt = 0; kt < 4; ++kt) {
            v16bf a = fragA(sX, m0, kt * 32, lane);
            v16bf bb = fragB(sWt, nt * 16, kt * 32, lane);
            acc = __builtin_amdgcn_wmma_f32_16x16x32_bf16(false, a, false, bb,
                                                          (short)0, acc, false, false);
        }
        const int n = nt * 16 + nlane;
        const float bias = b[n];
        float s = 0.f, s2 = 0.f;
#pragma unroll
        for (int r = 0; r < 8; ++r) {
            long long row = row0 + m0 + r + 8 * half;
            if (row < rows) {
                float v = acc[r] + bias;
                out[row * HID + n] = v;
                s += v; s2 += v * v;
            }
        }
        s  += __shfl_xor(s, 16, 32);     // combine the two half-wave row groups
        s2 += __shfl_xor(s2, 16, 32);
        if (lane < 16) { atomicAdd(&stats[n], s); atomicAdd(&stats[HID + n], s2); }
    }
}

// ---- small kernels ----------------------------------------------------------
__global__ __launch_bounds__(256) void zero_f32(float* p, long long n) {
    long long i = (long long)blockIdx.x * 256 + threadIdx.x;
    if (i < n) p[i] = 0.f;
}

__global__ __launch_bounds__(256) void enc_l1(const float* __restrict__ x, int din,
                                              const float* __restrict__ W, const float* __restrict__ b,
                                              float* __restrict__ h, long long rows) {
    long long i = (long long)blockIdx.x * 256 + threadIdx.x;
    if (i >= rows * HID) return;
    long long row = i >> 7; int c = (int)(i & 127);
    float s = b[c];
    for (int k = 0; k < din; ++k) s += x[row * din + k] * W[k * HID + c];
    h[i] = gelu_t(s);
}

__global__ __launch_bounds__(256) void bn_apply(float* __restrict__ xio, const float* __restrict__ stats,
                                                const float* __restrict__ g, const float* __restrict__ bt,
                                                long long rows) {
    long long i = (long long)blockIdx.x * 256 + threadIdx.x;
    if (i >= rows * HID) return;
    int c = (int)(i & (HID - 1));
    float inv = 1.0f / (float)rows;
    float mu  = stats[c] * inv;
    float var = stats[HID + c] * inv - mu * mu;           // biased variance
    xio[i] = g[c] * (xio[i] - mu) * rsqrtf(var + BN_EPS) + bt[c];
}

// m = gelu(nh[src]+eh); agg[dst] += m   (one wave per edge: 128ch as float4/lane)
__global__ __launch_bounds__(256) void edge_msg(const float* __restrict__ nh, const float* __restrict__ eh,
                                                const int* __restrict__ src, const int* __restrict__ dst,
                                                float* __restrict__ agg) {
    long long idx = (long long)blockIdx.x * 256 + threadIdx.x;
    if (idx >= (long long)NE * 32) return;
    long long e = idx >> 5;
    int c = (int)(idx & 31) * 4;
    int s = src[e], d = dst[e];
    if (e + 16 < NE)  // hide gather latency: prefetch a future source row
        __builtin_prefetch(nh + (long long)src[e + 16] * HID + c, 0, 1);
    float4 a = *(const float4*)(nh + (long long)s * HID + c);
    float4 b = *(const float4*)(eh + e * HID + c);
    float* p = agg + (long long)d * HID + c;
    atomicAdd(p + 0, gelu_t(a.x + b.x));
    atomicAdd(p + 1, gelu_t(a.y + b.y));
    atomicAdd(p + 2, gelu_t(a.z + b.z));
    atomicAdd(p + 3, gelu_t(a.w + b.w));
}

__global__ __launch_bounds__(256) void pool128(const float* __restrict__ nh, const int* __restrict__ batch,
                                               float* __restrict__ gh) {
    long long idx = (long long)blockIdx.x * 256 + threadIdx.x;
    if (idx >= (long long)NN * 32) return;
    long long v = idx >> 5;
    int c = (int)(idx & 31) * 4;
    int g = batch[v];
    float4 a = *(const float4*)(nh + v * HID + c);
    float* p = gh + (long long)g * HID + c;
    atomicAdd(p + 0, a.x); atomicAdd(p + 1, a.y);
    atomicAdd(p + 2, a.z); atomicAdd(p + 3, a.w);
}

// ---- host orchestration -----------------------------------------------------
extern "C" void kernel_launch(void* const* d_in, const int* in_sizes, int n_in,
                              void* d_out, int out_size, void* d_ws, size_t ws_size,
                              hipStream_t stream) {
    (void)in_sizes; (void)n_in; (void)out_size; (void)ws_size;
    // setup_inputs() insertion order: x, edge_attr, edge_index, batch_idx,
    // enc{nn,en,ne,ee,ng,eg}(W1,b1,W2,b2), bn{same keys}(g,b),
    // gin{n,e,g}(eps, mlp_n(W1,b1,W2,b2), mlp_e(W1,b1,W2,b2))
    const float* x   = (const float*)d_in[0];
    const float* ea  = (const float*)d_in[1];
    const int*   src = (const int*)d_in[2];
    const int*   dst = src + NE;
    const int*   bat = (const int*)d_in[3];

    const long long NHsz = (long long)NN * HID;
    const long long EHsz = (long long)NE * HID;
    float* ws    = (float*)d_ws;
    float* eh_a  = ws;
    float* eh_b  = eh_a + EHsz;
    float* nh_a  = eh_b + EHsz;
    float* nh_b  = nh_a + NHsz;
    float* agg   = nh_b + NHsz;
    float* stats = agg + NHsz;        // 256 floats: [sum(128), sumsq(128)]

    float* out_n = (float*)d_out;
    float* out_e = out_n + NHsz;
    float* out_g = out_e + EHsz;

    const int nodeTiles = (NN + 127) / 128;                    // 391
    const int edgeTiles = (NE + 127) / 128;                    // 6250
    const int nodeElemB = (int)((NHsz + 255) / 256);
    const int edgeElemB = (int)((EHsz + 255) / 256);
    const int msgB      = (int)(((long long)NE * 32 + 255) / 256);
    const int poolB     = (int)(((long long)NN * 32 + 255) / 256);

    for (int b = 0; b < 3; ++b) {
        const float* eNW1 = (const float*)d_in[4 + 8 * b + 0];
        const float* eNb1 = (const float*)d_in[4 + 8 * b + 1];
        const float* eNW2 = (const float*)d_in[4 + 8 * b + 2];
        const float* eNb2 = (const float*)d_in[4 + 8 * b + 3];
        const float* eEW1 = (const float*)d_in[8 + 8 * b + 0];
        const float* eEb1 = (const float*)d_in[8 + 8 * b + 1];
        const float* eEW2 = (const float*)d_in[8 + 8 * b + 2];
        const float* eEb2 = (const float*)d_in[8 + 8 * b + 3];
        const float* bnNg = (const float*)d_in[28 + 4 * b + 0];
        const float* bnNb = (const float*)d_in[28 + 4 * b + 1];
        const float* bnEg = (const float*)d_in[30 + 4 * b + 0];
        const float* bnEb = (const float*)d_in[30 + 4 * b + 1];
        const float* eps  = (const float*)d_in[40 + 9 * b];
        const float* gNW1 = (const float*)d_in[41 + 9 * b + 0];
        const float* gNb1 = (const float*)d_in[41 + 9 * b + 1];
        const float* gNW2 = (const float*)d_in[41 + 9 * b + 2];
        const float* gNb2 = (const float*)d_in[41 + 9 * b + 3];
        const float* gEW1 = (const float*)d_in[45 + 9 * b + 0];
        const float* gEb1 = (const float*)d_in[45 + 9 * b + 1];
        const float* gEW2 = (const float*)d_in[45 + 9 * b + 2];
        const float* gEb2 = (const float*)d_in[45 + 9 * b + 3];

        // encoders + training-mode BN (stats fused into WMMA linear epilogue)
        enc_l1<<<nodeElemB, 256, 0, stream>>>(x, 9, eNW1, eNb1, nh_b, NN);
        zero_f32<<<1, 256, 0, stream>>>(stats, 256);
        linear128_stats<<<nodeTiles, 256, 0, stream>>>(nh_b, eNW2, eNb2, nh_a, stats, NN);
        bn_apply<<<nodeElemB, 256, 0, stream>>>(nh_a, stats, bnNg, bnNb, NN);

        enc_l1<<<edgeElemB, 256, 0, stream>>>(ea, 5, eEW1, eEb1, eh_b, NE);
        zero_f32<<<1, 256, 0, stream>>>(stats, 256);
        linear128_stats<<<edgeTiles, 256, 0, stream>>>(eh_b, eEW2, eEb2, eh_a, stats, NE);
        bn_apply<<<edgeElemB, 256, 0, stream>>>(eh_a, stats, bnEg, bnEb, NE);

        float* nh_cur = nh_a; float* nh_nxt = nh_b;
        float* eh_cur = eh_a; float* eh_nxt = eh_b;

        for (int d = 0; d < GDEPTH; ++d) {
            zero_f32<<<nodeElemB, 256, 0, stream>>>(agg, NHsz);
            edge_msg<<<msgB, 256, 0, stream>>>(nh_cur, eh_cur, src, dst, agg);

            float* nout = (d == GDEPTH - 1 && b == 0) ? out_n : nh_nxt;
            mlp128_fused<<<nodeTiles, 256, 0, stream>>>(nh_cur, agg, nullptr, nullptr,
                                                        eps, 1, gNW1, gNb1, gNW2, gNb2, nout, NN);
            // final-depth edge MLP is dead work for branches 'n' (b==0) and 'g' (b==2)
            bool needEdge = !(d == GDEPTH - 1 && (b == 0 || b == 2));
            if (needEdge) {
                float* eout = (d == GDEPTH - 1 && b == 1) ? out_e : eh_nxt;
                mlp128_fused<<<edgeTiles, 256, 0, stream>>>(eh_cur, nout, src, dst,
                                                            nullptr, 2, gEW1, gEb1, gEW2, gEb2, eout, NE);
                if (eout == eh_nxt) { float* t = eh_cur; eh_cur = eh_nxt; eh_nxt = t; }
            }
            if (nout == nh_nxt) { float* t = nh_cur; nh_cur = nh_nxt; nh_nxt = t; }
            else                { nh_cur = nout; }
        }
        if (b == 2) {
            zero_f32<<<(NGRAPH * HID + 255) / 256, 256, 0, stream>>>(out_g, (long long)NGRAPH * HID);
            pool128<<<poolB, 256, 0, stream>>>(nh_cur, bat, out_g);
        }
    }
}
